// MMDLoss_81080392613941
// MI455X (gfx1250) — matrix operand order repaired and checked by reading, not compiled, not run on hardware
//
#include <hip/hip_runtime.h>
#include <hip/hip_bf16.h>

// ---------------------------------------------------------------------------
// MMD loss via compensated-bf16 WMMA Gram matrices on gfx1250 (MI455X).
//   d2(i,j) = ||x_i||^2 + ||y_j||^2 - 2 * dot(x_i, y_j)
//   dot = xh*yh + xh*yl + xl*yh  (hi/lo bf16 split, ~2^-17 rel err)
// Fast path: bf16 panels pre-split in workspace, double-buffered LDS slabs
// filled with GLOBAL_LOAD_ASYNC_TO_LDS_B128 (ASYNCcnt), 24 wmma / wave / slab.
// Fallback path (small ws): convert f32->bf16 inside the Gram kernel.
// ---------------------------------------------------------------------------

typedef __attribute__((ext_vector_type(16))) __bf16 bf16x16;
typedef __attribute__((ext_vector_type(8)))  __bf16 bf16x8;
typedef __attribute__((ext_vector_type(4)))  __bf16 bf16x4;
typedef __attribute__((ext_vector_type(8)))  float  f32x8;
typedef __attribute__((ext_vector_type(4)))  int    i32x4;

#define NROWS 8192
#define KDIM  256
#define TILE  128
#define KT    32
#define LDSK  40   // padded LDS k-stride (80B rows: 16B aligned, conflict-free)
#define NSLAB (KDIM / KT)

union V16 { bf16x16 v; bf16x8 h[2]; };
union V8  { bf16x8  v; __bf16 e[8]; };
union V4  { bf16x4  v; __bf16 e[4]; };

// ---- async global->LDS copy of 16 bf16 (two b128), with sync fallback ----
__device__ __forceinline__ void copy16_lds(__bf16* dst, const __bf16* src) {
#if __has_builtin(__builtin_amdgcn_global_load_async_to_lds_b128)
  // Builtin expects int4 pointers in global(1)/local(3) address spaces.
  // Integer round-trip produces the AS-qualified pointers (generic LDS
  // addresses carry the LDS offset in their low 32 bits).
  auto* g0 = reinterpret_cast<__attribute__((address_space(1))) i32x4*>(
      reinterpret_cast<uintptr_t>(src));
  auto* l0 = reinterpret_cast<__attribute__((address_space(3))) i32x4*>(
      reinterpret_cast<uintptr_t>(dst));
  auto* g1 = reinterpret_cast<__attribute__((address_space(1))) i32x4*>(
      reinterpret_cast<uintptr_t>(src + 8));
  auto* l1 = reinterpret_cast<__attribute__((address_space(3))) i32x4*>(
      reinterpret_cast<uintptr_t>(dst + 8));
  __builtin_amdgcn_global_load_async_to_lds_b128(g0, l0, 0, 0);
  __builtin_amdgcn_global_load_async_to_lds_b128(g1, l1, 0, 0);
#else
  *(bf16x8*)dst       = *(const bf16x8*)src;
  *(bf16x8*)(dst + 8) = *(const bf16x8*)(src + 8);
#endif
}

__device__ __forceinline__ void wait_async0() {
#if __has_builtin(__builtin_amdgcn_global_load_async_to_lds_b128)
#if __has_builtin(__builtin_amdgcn_s_wait_asynccnt)
  __builtin_amdgcn_s_wait_asynccnt(0);
#else
  asm volatile("s_wait_asynccnt 0x0" ::: "memory");
#endif
#endif
}

// ---------------------------------------------------------------------------
// Pass 1: row squared norms for both matrices; zero the 3 accumulators.
// ---------------------------------------------------------------------------
__global__ __launch_bounds__(256) void mmd_sqnorm(const float* __restrict__ Np,
                                                  const float* __restrict__ Rp,
                                                  float* __restrict__ sq,
                                                  float* __restrict__ acc) {
  const int t = threadIdx.x;
  if (blockIdx.x == 0 && t < 3) acc[t] = 0.0f;

  const int wave = t >> 5;
  const int lane = t & 31;
  const int row  = blockIdx.x * 8 + wave;          // 0..16383
  const float* src = (row < NROWS) ? (Np + (size_t)row * KDIM)
                                   : (Rp + (size_t)(row - NROWS) * KDIM);
  const float4* s4 = (const float4*)src;
  float4 a = s4[lane];
  float4 b = s4[lane + 32];
  float s = a.x*a.x + a.y*a.y + a.z*a.z + a.w*a.w
          + b.x*b.x + b.y*b.y + b.z*b.z + b.w*b.w;
  #pragma unroll
  for (int off = 16; off > 0; off >>= 1) s += __shfl_xor(s, off, 32);
  if (lane == 0) sq[row] = s;
}

// ---------------------------------------------------------------------------
// Pass 1b (fast path): one-shot f32 -> bf16 hi/lo split into panels.
// ---------------------------------------------------------------------------
__global__ __launch_bounds__(256) void mmd_split(const float* __restrict__ Np,
                                                 const float* __restrict__ Rp,
                                                 __bf16* __restrict__ Nh,
                                                 __bf16* __restrict__ Nl,
                                                 __bf16* __restrict__ Rh,
                                                 __bf16* __restrict__ Rl) {
  const size_t idx = (size_t)blockIdx.x * 256 + threadIdx.x;  // float4 index
  {
    float4 f = ((const float4*)Np)[idx];
    float v[4] = {f.x, f.y, f.z, f.w};
    V4 h, l;
    #pragma unroll
    for (int i = 0; i < 4; ++i) {
      __bf16 hb = (__bf16)v[i];
      h.e[i] = hb; l.e[i] = (__bf16)(v[i] - (float)hb);
    }
    *(bf16x4*)&Nh[idx * 4] = h.v;
    *(bf16x4*)&Nl[idx * 4] = l.v;
  }
  {
    float4 f = ((const float4*)Rp)[idx];
    float v[4] = {f.x, f.y, f.z, f.w};
    V4 h, l;
    #pragma unroll
    for (int i = 0; i < 4; ++i) {
      __bf16 hb = (__bf16)v[i];
      h.e[i] = hb; l.e[i] = (__bf16)(v[i] - (float)hb);
    }
    *(bf16x4*)&Rh[idx * 4] = h.v;
    *(bf16x4*)&Rl[idx * 4] = l.v;
  }
}

// ---- 24 compensated WMMAs on one staged k-slab ----
__device__ __forceinline__ void compute_slab(const __bf16* Ah, const __bf16* Al,
                                             const __bf16* Bh, const __bf16* Bl,
                                             int m, int hi, int wr, int wc,
                                             f32x8 (&accv)[2][4]) {
  // A 16x32 bf16 frag: lanes 0-15 hold K 0..7 & 16..23 of row m; lanes 16-31: K 8..15 & 24..31
  V16 aH[2], aL[2], bH[4], bL[4];
  #pragma unroll
  for (int r = 0; r < 2; ++r) {
    const int o0 = (wr + r * 16 + m) * LDSK + hi * 8;
    aH[r].h[0] = *(const bf16x8*)&Ah[o0];
    aH[r].h[1] = *(const bf16x8*)&Ah[o0 + 16];
    aL[r].h[0] = *(const bf16x8*)&Al[o0];
    aL[r].h[1] = *(const bf16x8*)&Al[o0 + 16];
  }
  // B 32x16 bf16 frag: lanes 0-15 K 0..7 & 16..23 of col m; lanes 16-31: K 8..15 & 24..31
  #pragma unroll
  for (int c = 0; c < 4; ++c) {
    const int o0 = (wc + c * 16 + m) * LDSK + hi * 8;
    bH[c].h[0] = *(const bf16x8*)&Bh[o0];
    bH[c].h[1] = *(const bf16x8*)&Bh[o0 + 16];
    bL[c].h[0] = *(const bf16x8*)&Bl[o0];
    bL[c].h[1] = *(const bf16x8*)&Bl[o0 + 16];
  }
  #pragma unroll
  for (int r = 0; r < 2; ++r)
    #pragma unroll
    for (int c = 0; c < 4; ++c) {
      accv[r][c] = __builtin_amdgcn_wmma_f32_16x16x32_bf16(
          false, aH[r].v, false, bH[c].v, (short)0, accv[r][c], false, false);
      accv[r][c] = __builtin_amdgcn_wmma_f32_16x16x32_bf16(
          false, aH[r].v, false, bL[c].v, (short)0, accv[r][c], false, false);
      accv[r][c] = __builtin_amdgcn_wmma_f32_16x16x32_bf16(
          false, aL[r].v, false, bH[c].v, (short)0, accv[r][c], false, false);
    }
}

// ---------------------------------------------------------------------------
// Pass 2: one 128x128 Gram tile per block; blockIdx.z selects pair:
//   z=0: (N,N)  z=1: (R,R)  z=2: (N,R)
// PRE=true : read pre-split bf16 panels, async copy to double-buffered LDS.
// PRE=false: read f32, convert to bf16 hi/lo while staging (small-ws path).
// ---------------------------------------------------------------------------
template <bool PRE>
__global__ __launch_bounds__(256) void mmd_gram(const float* __restrict__ Np,
                                                const float* __restrict__ Rp,
                                                const __bf16* __restrict__ NhP,
                                                const __bf16* __restrict__ NlP,
                                                const __bf16* __restrict__ RhP,
                                                const __bf16* __restrict__ RlP,
                                                const float* __restrict__ sq,
                                                float* __restrict__ acc) {
  const int z = blockIdx.z;
  const float*  Af = (z == 1) ? Rp : Np;
  const float*  Bf = (z == 0) ? Np : Rp;
  const __bf16* pAh = (z == 1) ? RhP : NhP;
  const __bf16* pAl = (z == 1) ? RlP : NlP;
  const __bf16* pBh = (z == 0) ? NhP : RhP;
  const __bf16* pBl = (z == 0) ? NlP : RlP;
  const float* sqA = (z == 1) ? (sq + NROWS) : sq;
  const float* sqB = (z == 0) ? sq : (sq + NROWS);

  const int rowBase = blockIdx.y * TILE;
  const int colBase = blockIdx.x * TILE;

  __shared__ __align__(16) __bf16 ldsAh[2 * TILE * LDSK];
  __shared__ __align__(16) __bf16 ldsAl[2 * TILE * LDSK];
  __shared__ __align__(16) __bf16 ldsBh[2 * TILE * LDSK];
  __shared__ __align__(16) __bf16 ldsBl[2 * TILE * LDSK];
  __shared__ float red[8];

  const int t    = threadIdx.x;
  const int lane = t & 31;
  const int wave = t >> 5;
  const int m    = lane & 15;
  const int hi   = lane >> 4;
  const int wr   = (wave >> 1) * 32;   // wave row offset in tile
  const int wc   = (wave & 1) * 64;    // wave col offset in tile

  // staging assignment: thread -> (row, 16-element half of the 32-wide k slab)
  const int lrow  = t >> 1;
  const int lhalf = (t & 1) << 4;
  const int so    = lrow * LDSK + lhalf;

  f32x8 accv[2][4];
  #pragma unroll
  for (int r = 0; r < 2; ++r)
    #pragma unroll
    for (int c = 0; c < 4; ++c)
      #pragma unroll
      for (int g = 0; g < 8; ++g) accv[r][c][g] = 0.0f;

  auto stage = [&](int kt, int buf) {
    const int bo = buf * TILE * LDSK;
    if constexpr (PRE) {
      const size_t ga = (size_t)(rowBase + lrow) * KDIM + kt + lhalf;
      const size_t gb = (size_t)(colBase + lrow) * KDIM + kt + lhalf;
      copy16_lds(&ldsAh[bo + so], pAh + ga);
      copy16_lds(&ldsAl[bo + so], pAl + ga);
      copy16_lds(&ldsBh[bo + so], pBh + gb);
      copy16_lds(&ldsBl[bo + so], pBl + gb);
    } else {
      const float4* sa = (const float4*)(Af + (size_t)(rowBase + lrow) * KDIM + kt + lhalf);
      const float4* sb = (const float4*)(Bf + (size_t)(colBase + lrow) * KDIM + kt + lhalf);
      float va[16], vb[16];
      #pragma unroll
      for (int q = 0; q < 4; ++q) {
        float4 fa = sa[q], fb = sb[q];
        va[4*q+0] = fa.x; va[4*q+1] = fa.y; va[4*q+2] = fa.z; va[4*q+3] = fa.w;
        vb[4*q+0] = fb.x; vb[4*q+1] = fb.y; vb[4*q+2] = fb.z; vb[4*q+3] = fb.w;
      }
      V8 ah0, ah1, al0, al1, bh0, bh1, bl0, bl1;
      #pragma unroll
      for (int i = 0; i < 8; ++i) {
        __bf16 h;
        h = (__bf16)va[i];     ah0.e[i] = h; al0.e[i] = (__bf16)(va[i]     - (float)h);
        h = (__bf16)va[i + 8]; ah1.e[i] = h; al1.e[i] = (__bf16)(va[i + 8] - (float)h);
        h = (__bf16)vb[i];     bh0.e[i] = h; bl0.e[i] = (__bf16)(vb[i]     - (float)h);
        h = (__bf16)vb[i + 8]; bh1.e[i] = h; bl1.e[i] = (__bf16)(vb[i + 8] - (float)h);
      }
      *(bf16x8*)&ldsAh[bo + so] = ah0.v;  *(bf16x8*)&ldsAh[bo + so + 8] = ah1.v;
      *(bf16x8*)&ldsAl[bo + so] = al0.v;  *(bf16x8*)&ldsAl[bo + so + 8] = al1.v;
      *(bf16x8*)&ldsBh[bo + so] = bh0.v;  *(bf16x8*)&ldsBh[bo + so + 8] = bh1.v;
      *(bf16x8*)&ldsBl[bo + so] = bl0.v;  *(bf16x8*)&ldsBl[bo + so + 8] = bl1.v;
    }
  };

  // prologue: stage slab 0 into buffer 0
  stage(0, 0);
  if constexpr (PRE) wait_async0();
  __syncthreads();

  #pragma unroll
  for (int s = 0; s < NSLAB; ++s) {
    const int cur = s & 1;
    if (s + 1 < NSLAB) stage((s + 1) * KT, cur ^ 1);  // overlap copy w/ compute
    const int bo = cur * TILE * LDSK;
    compute_slab(&ldsAh[bo], &ldsAl[bo], &ldsBh[bo], &ldsBl[bo],
                 m, hi, wr, wc, accv);
    if constexpr (PRE) { if (s + 1 < NSLAB) wait_async0(); }
    __syncthreads();
  }

  // ---- epilogue: exp(-max(d2,0)), block reduce, one atomic per block ----
  // C/D layout: VGPR g, lanes 0-15 -> (M=g, N=lane); lanes 16-31 -> (M=8+g, N=lane-16)
  float lsum = 0.0f;
  #pragma unroll
  for (int r = 0; r < 2; ++r) {
    const int i0 = rowBase + wr + r * 16 + hi * 8;
    float sa[8];
    #pragma unroll
    for (int g = 0; g < 8; ++g) sa[g] = sqA[i0 + g];
    #pragma unroll
    for (int c = 0; c < 4; ++c) {
      const float sb = sqB[colBase + wc + c * 16 + m];
      #pragma unroll
      for (int g = 0; g < 8; ++g) {
        float d2 = sa[g] + sb - 2.0f * accv[r][c][g];
        d2 = fmaxf(d2, 0.0f);
        lsum += __expf(-d2);   // GAMMA = 1.0
      }
    }
  }
  #pragma unroll
  for (int off = 16; off > 0; off >>= 1) lsum += __shfl_xor(lsum, off, 32);
  if (lane == 0) red[wave] = lsum;
  __syncthreads();
  if (t == 0) {
    float s = 0.0f;
    #pragma unroll
    for (int w = 0; w < 8; ++w) s += red[w];
    atomicAdd(&acc[z], s);
  }
}

// ---------------------------------------------------------------------------
// Pass 3: combine.  mmd = mean(xx) + mean(yy) - 2*mean(xy); out = sqrt(mmd)
// ---------------------------------------------------------------------------
__global__ void mmd_final(const float* __restrict__ acc, float* __restrict__ out) {
  if (threadIdx.x == 0) {
    const float s = 1.0f / (8192.0f * 8192.0f);  // exact power of two
    float mmd = acc[0] * s + acc[1] * s - 2.0f * (acc[2] * s);
    out[0] = sqrtf(fmaxf(mmd, 0.0f));
  }
}

extern "C" void kernel_launch(void* const* d_in, const int* in_sizes, int n_in,
                              void* d_out, int out_size, void* d_ws, size_t ws_size,
                              hipStream_t stream) {
  (void)in_sizes; (void)n_in; (void)out_size;
  const float* Np = (const float*)d_in[0];
  const float* Rp = (const float*)d_in[1];
  float* ws  = (float*)d_ws;
  float* sq  = ws;              // 16384 floats: ||n_i||^2 then ||r_i||^2
  float* acc = ws + 2 * NROWS;  // 3 floats: sum_xx, sum_yy, sum_xy

  const size_t panel_elems = (size_t)NROWS * KDIM;           // 2,097,152 bf16 each
  __bf16* Nh = (__bf16*)(ws + 16640);                        // 256B-aligned offset
  __bf16* Nl = Nh + panel_elems;
  __bf16* Rh = Nl + panel_elems;
  __bf16* Rl = Rh + panel_elems;
  const size_t need = 16640 * sizeof(float) + 4 * panel_elems * sizeof(__bf16);

  mmd_sqnorm<<<(2 * NROWS) / 8, 256, 0, stream>>>(Np, Rp, sq, acc);

  const dim3 grid(NROWS / TILE, NROWS / TILE, 3);
  if (ws_size >= need) {
    mmd_split<<<(int)(panel_elems / 4 / 256), 256, 0, stream>>>(Np, Rp, Nh, Nl, Rh, Rl);
    mmd_gram<true><<<grid, 256, 0, stream>>>(Np, Rp, Nh, Nl, Rh, Rl, sq, acc);
  } else {
    mmd_gram<false><<<grid, 256, 0, stream>>>(Np, Rp, Nh, Nl, Rh, Rl, sq, acc);
  }

  mmd_final<<<1, 32, 0, stream>>>(acc, (float*)d_out);
}
